// MLPGate2_16149077033387
// MI455X (gfx1250) — matrix operand order repaired and checked by reading, not compiled, not run on hardware
//
#include <hip/hip_runtime.h>
#include <hip/hip_bf16.h>

typedef _Float16 f16;
typedef __attribute__((ext_vector_type(16))) _Float16 v16h;
typedef __attribute__((ext_vector_type(8)))  _Float16 v8h;
typedef __attribute__((ext_vector_type(8)))  float    v8f;
typedef __attribute__((ext_vector_type(4)))  float    v4f;
typedef __attribute__((ext_vector_type(4)))  unsigned int u32x4;
typedef __attribute__((ext_vector_type(4)))  int      i32x4;
typedef __attribute__((ext_vector_type(8)))  int      i32x8;

#define NN 2048
#define EE 16384
#define SS 256
#define DH 128
#define H2 256
#define G3 768

#if defined(__has_include)
#if __has_include(<hip/amd_detail/amd_gfx1250_TDM.h>)
#define TDM_SIX_ARGS 1
#endif
#endif

// ---------------------------------------------------------------------------
// WMMA helpers
// ---------------------------------------------------------------------------
__device__ __forceinline__ v8f wmma_f16(v16h a, v16h b, v8f c) {
  return __builtin_amdgcn_wmma_f32_16x16x32_f16(false, a, false, b, (short)0, c,
                                                false, false);
}

// A fragment (16x32) from an f32 row: 4x float4 (16B-aligned), convert to f16.
__device__ __forceinline__ v16h load_a_f32(const float* __restrict__ Arow, int kA) {
  v4f x0 = *(const v4f*)(Arow + kA);
  v4f x1 = *(const v4f*)(Arow + kA + 4);
  v4f x2 = *(const v4f*)(Arow + kA + 8);
  v4f x3 = *(const v4f*)(Arow + kA + 12);
  v16h a;
#pragma unroll
  for (int i = 0; i < 4; ++i) {
    a[i]      = (_Float16)x0[i];
    a[4 + i]  = (_Float16)x1[i];
    a[8 + i]  = (_Float16)x2[i];
    a[12 + i] = (_Float16)x3[i];
  }
  return a;
}

// B fragment from n-major (nk) f16 in global: 16 contiguous halves = 2 x b128.
__device__ __forceinline__ v16h load_b16(const f16* __restrict__ p) {
  v8h lo = *(const v8h*)(p);
  v8h hi = *(const v8h*)(p + 8);
  return __builtin_shufflevector(lo, hi, 0, 1, 2, 3, 4, 5, 6, 7, 8, 9, 10, 11,
                                 12, 13, 14, 15);
}

// ---------------------------------------------------------------------------
// Generic WMMA GEMM, B pre-transposed to n-major f16 (B[n*ldb + k]).
// ---------------------------------------------------------------------------
template <int KT>
__global__ void gemm_wmma(const float* __restrict__ A, int lda,
                          const f16* __restrict__ B, int ldb,
                          const float* __restrict__ bias,
                          float* __restrict__ C, int ldc, int M, int Nc) {
  const int wave   = threadIdx.x >> 5;
  const int lane   = threadIdx.x & 31;
  const int laneHi = lane >> 4;
  const int laneLo = lane & 15;
  const int row0 = (blockIdx.y * 8 + wave) * 16;
  const int col0 = blockIdx.x * 64;
  if (row0 >= M || col0 >= Nc) return;

  v8f acc0 = {}, acc1 = {}, acc2 = {}, acc3 = {};
  const float* Arow = A + (size_t)(row0 + laneLo) * lda;
  const f16* B0 = B + (size_t)(col0 + laneLo) * ldb;
  const f16* B1 = B0 + (size_t)16 * ldb;
  const f16* B2 = B0 + (size_t)32 * ldb;
  const f16* B3 = B0 + (size_t)48 * ldb;

#pragma unroll
  for (int k0 = 0; k0 < KT; k0 += 32) {
    v16h a = load_a_f32(Arow, k0 + laneHi * 16);
    const int kb = k0 + laneHi * 16;
    acc0 = wmma_f16(a, load_b16(B0 + kb), acc0);
    acc1 = wmma_f16(a, load_b16(B1 + kb), acc1);
    acc2 = wmma_f16(a, load_b16(B2 + kb), acc2);
    acc3 = wmma_f16(a, load_b16(B3 + kb), acc3);
  }

  const v8f accs[4] = {acc0, acc1, acc2, acc3};
#pragma unroll
  for (int t = 0; t < 4; ++t) {
    const int col = col0 + t * 16 + laneLo;
    const float bv = bias[col];
#pragma unroll
    for (int r = 0; r < 8; ++r) {
      const int row = row0 + laneHi * 8 + r;
      C[(size_t)row * ldc + col] = accs[t][r] + bv;
    }
  }
}

// ---------------------------------------------------------------------------
// K=256 GEMM with Tensor-Data-Mover staging: B panel (64 cols x 256 K = 32KB)
// DMA'd to LDS once per block; fragments served from LDS.
// ---------------------------------------------------------------------------
__device__ __forceinline__ void tdm_load_1d(const f16* gsrc, unsigned nelem) {
  unsigned long long ga = (unsigned long long)gsrc;
  u32x4 g0;
  g0[0] = 1u;                                   // count = 1 (valid descriptor)
  g0[1] = 0u;                                   // lds_addr = 0 (block LDS base)
  g0[2] = (unsigned)(ga & 0xffffffffu);         // global_addr[31:0]
  g0[3] = (unsigned)((ga >> 32) & 0x01ffffffu) | 0x80000000u;  // addr hi | type=2
  i32x8 g1;
  g1[0] = 0x00010000;                           // data_size = 2B
  g1[1] = (int)((nelem & 0xffffu) << 16);       // tensor_dim0 low16
  g1[2] = (int)(((nelem >> 16) & 0xffffu)) | 0x00010000;  // dim0 hi | tensor_dim1=1
  g1[3] = (int)((nelem & 0xffffu) << 16);       // tile_dim0
  g1[4] = 1;                                    // tile_dim1 = 1
  g1[5] = (int)nelem;                           // tensor_dim0_stride
  g1[6] = 0;
  g1[7] = 0;
  i32x4 z4 = {0, 0, 0, 0};
#ifdef TDM_SIX_ARGS
  i32x8 z8 = {0, 0, 0, 0, 0, 0, 0, 0};
  __builtin_amdgcn_tensor_load_to_lds(g0, g1, z4, z4, z8, 0);
#else
  __builtin_amdgcn_tensor_load_to_lds(g0, g1, z4, z4, 0);
#endif
}

__global__ void gemm_wmma_tdm(const float* __restrict__ A, int lda,
                              const f16* __restrict__ Bt,   // Nc x 256 n-major
                              const float* __restrict__ bias,
                              float* __restrict__ C, int ldc, int M) {
  __shared__ f16 Bs[64 * 256];  // 32 KB panel
  const int wave   = threadIdx.x >> 5;
  const int lane   = threadIdx.x & 31;
  const int laneHi = lane >> 4;
  const int laneLo = lane & 15;
  const int row0 = (blockIdx.y * 8 + wave) * 16;
  const int col0 = blockIdx.x * 64;

  // Opaque zero: the optimizer cannot prove the guarded store below is dead,
  // so it must assume Bs may be written (the TDM builtin's LDS write is not
  // modeled by the compiler, which otherwise folds Bs loads to undef).
  int opaque_zero;
  asm volatile("s_mov_b32 %0, 0" : "=s"(opaque_zero));
  if (opaque_zero) Bs[threadIdx.x] = (f16)0.f;

  if (wave == 0) {
    tdm_load_1d(Bt + (size_t)col0 * 256, 64 * 256);
    __builtin_amdgcn_s_wait_tensorcnt(0);
  }
  __syncthreads();
  asm volatile("" ::: "memory");  // LDS now holds the TDM-written panel

  v8f acc0 = {}, acc1 = {}, acc2 = {}, acc3 = {};
  const float* Arow = A + (size_t)(row0 + laneLo) * lda;
  const int b0 = laneLo * 256;

#pragma unroll
  for (int k0 = 0; k0 < 256; k0 += 32) {
    v16h a = load_a_f32(Arow, k0 + laneHi * 16);
    const int kb = k0 + laneHi * 16;
#define LDS_FRAG(ofs)                                                        \
  __builtin_shufflevector(*(const v8h*)&Bs[(ofs) + kb],                      \
                          *(const v8h*)&Bs[(ofs) + kb + 8], 0, 1, 2, 3, 4, 5,\
                          6, 7, 8, 9, 10, 11, 12, 13, 14, 15)
    acc0 = wmma_f16(a, LDS_FRAG(b0), acc0);
    acc1 = wmma_f16(a, LDS_FRAG(b0 + 16 * 256), acc1);
    acc2 = wmma_f16(a, LDS_FRAG(b0 + 32 * 256), acc2);
    acc3 = wmma_f16(a, LDS_FRAG(b0 + 48 * 256), acc3);
#undef LDS_FRAG
  }

  const v8f accs[4] = {acc0, acc1, acc2, acc3};
#pragma unroll
  for (int t = 0; t < 4; ++t) {
    const int col = col0 + t * 16 + laneLo;
    const float bv = bias[col];
#pragma unroll
    for (int r = 0; r < 8; ++r) {
      const int row = row0 + laneHi * 8 + r;
      C[(size_t)row * ldc + col] = accs[t][r] + bv;
    }
  }
}

// ---------------------------------------------------------------------------
// Edge aggregation: m_e = relu([h[g[e]], ea[e]] @ W + b); agg[s[e]] += m_e.
// ---------------------------------------------------------------------------
__global__ void aggr_wmma(const float* __restrict__ h,
                          const f16* __restrict__ Wt,  // 256 x 288 n-major
                          const float* __restrict__ bias,
                          const float* __restrict__ ea,
                          const int* __restrict__ gIdx,
                          const int* __restrict__ sIdx,
                          float* __restrict__ agg) {
  const int wave   = threadIdx.x >> 5;
  const int lane   = threadIdx.x & 31;
  const int laneHi = lane >> 4;
  const int laneLo = lane & 15;
  const int row0 = (blockIdx.y * 8 + wave) * 16;
  const int col0 = blockIdx.x * 64;
  if (row0 >= EE) return;

  const int eA = row0 + laneLo;
  const float* hr = h + (size_t)gIdx[eA] * H2;
  const float* er = ea + (size_t)eA * 16;
  const f16* B0 = Wt + (size_t)(col0 + laneLo) * 288;
  const f16* B1 = B0 + (size_t)16 * 288;
  const f16* B2 = B0 + (size_t)32 * 288;
  const f16* B3 = B0 + (size_t)48 * 288;

  v8f acc0 = {}, acc1 = {}, acc2 = {}, acc3 = {};
#pragma unroll
  for (int k0 = 0; k0 < 256; k0 += 32) {
    v16h a = load_a_f32(hr, k0 + laneHi * 16);
    const int kb = k0 + laneHi * 16;
    acc0 = wmma_f16(a, load_b16(B0 + kb), acc0);
    acc1 = wmma_f16(a, load_b16(B1 + kb), acc1);
    acc2 = wmma_f16(a, load_b16(B2 + kb), acc2);
    acc3 = wmma_f16(a, load_b16(B3 + kb), acc3);
  }
  {  // K block [256, 288): edge attributes + zero pad
    v4f e0 = *(const v4f*)(er);
    v4f e1 = *(const v4f*)(er + 4);
    v4f e2 = *(const v4f*)(er + 8);
    v4f e3 = *(const v4f*)(er + 12);
    v16h a;
#pragma unroll
    for (int i = 0; i < 4; ++i) {
      a[i]      = laneHi ? (_Float16)0.f : (_Float16)e0[i];
      a[4 + i]  = laneHi ? (_Float16)0.f : (_Float16)e1[i];
      a[8 + i]  = laneHi ? (_Float16)0.f : (_Float16)e2[i];
      a[12 + i] = laneHi ? (_Float16)0.f : (_Float16)e3[i];
    }
    const int kb = 256 + laneHi * 16;
    acc0 = wmma_f16(a, load_b16(B0 + kb), acc0);
    acc1 = wmma_f16(a, load_b16(B1 + kb), acc1);
    acc2 = wmma_f16(a, load_b16(B2 + kb), acc2);
    acc3 = wmma_f16(a, load_b16(B3 + kb), acc3);
  }

  const v8f accs[4] = {acc0, acc1, acc2, acc3};
#pragma unroll
  for (int t = 0; t < 4; ++t) {
    const int col = col0 + t * 16 + laneLo;
    const float bv = bias[col];
#pragma unroll
    for (int r = 0; r < 8; ++r) {
      const int e = row0 + laneHi * 8 + r;
      float v = fmaxf(accs[t][r] + bv, 0.f);
      atomicAdd(&agg[(size_t)sIdx[e] * H2 + col], v);
    }
  }
}

// ---------------------------------------------------------------------------
// Elementwise / small kernels
// ---------------------------------------------------------------------------
__global__ void fill_zero(float* __restrict__ p, int n) {
  int i = blockIdx.x * 256 + threadIdx.x;
  if (i < n) p[i] = 0.f;
}

__global__ void f32_to_f16_k(const float* __restrict__ s, f16* __restrict__ d, int n) {
  int i = blockIdx.x * 256 + threadIdx.x;
  if (i < n) d[i] = (f16)s[i];
}

// dst[n*Kpad + k] = (k < K) ? src[k*N + n] : 0   (kn f32 -> nk f16, K padded)
__global__ void transpose_cvt(const float* __restrict__ src, f16* __restrict__ dst,
                              int K, int N, int Kpad) {
  int idx = blockIdx.x * 256 + threadIdx.x;
  if (idx >= N * Kpad) return;
  int n = idx / Kpad, k = idx % Kpad;
  dst[idx] = (k < K) ? (f16)src[(size_t)k * N + n] : (f16)0.f;
}

__global__ void gru_pointwise(float* __restrict__ h, const float* __restrict__ gi,
                              const float* __restrict__ gh) {
  int idx = blockIdx.x * 256 + threadIdx.x;
  if (idx >= NN * H2) return;
  int node = idx >> 8, c = idx & 255;
  const float* gir = gi + (size_t)node * G3;
  const float* ghr = gh + (size_t)node * G3;
  float r  = 1.f / (1.f + expf(-(gir[c] + ghr[c])));
  float z  = 1.f / (1.f + expf(-(gir[256 + c] + ghr[256 + c])));
  float nn = tanhf(gir[512 + c] + r * ghr[512 + c]);
  float hv = h[idx];
  h[idx] = (1.f - z) * nn + z * hv;
}

// score[n] = dot(tanh(h[n] @ W1 + b1), v)   (block = node, 128 threads)
__global__ void pool_score(const float* __restrict__ h, int ldh,
                           const float* __restrict__ W1, const float* __restrict__ b1,
                           const float* __restrict__ v, float* __restrict__ score) {
  __shared__ float xs[128];
  __shared__ float red[128];
  int n = blockIdx.x, c = threadIdx.x;
  xs[c] = h[(size_t)n * ldh + c];
  __syncthreads();
  float acc = b1[c];
  for (int k = 0; k < 128; ++k) acc += xs[k] * W1[k * 128 + c];
  red[c] = tanhf(acc) * v[c];
  __syncthreads();
  for (int s2 = 64; s2 > 0; s2 >>= 1) {
    if (c < s2) red[c] += red[c + s2];
    __syncthreads();
  }
  if (c == 0) score[n] = red[0];
}

// Common softmax mass over n <= 2016 (mask always selects hA there).
__global__ void pool_common(const float* __restrict__ sA,
                            const float* __restrict__ hA, int ldh,
                            float* __restrict__ cvec, float* __restrict__ cms) {
  const int NC = 2017;
  __shared__ float red[128];
  int t = threadIdx.x;
  float mx = -3.0e38f;
  for (int n = t; n < NC; n += 128) mx = fmaxf(mx, sA[n]);
  red[t] = mx;
  __syncthreads();
  for (int s2 = 64; s2 > 0; s2 >>= 1) {
    if (t < s2) red[t] = fmaxf(red[t], red[t + s2]);
    __syncthreads();
  }
  float M = red[0];
  __syncthreads();
  float sm = 0.f;
  for (int n = t; n < NC; n += 128) sm += expf(sA[n] - M);
  red[t] = sm;
  __syncthreads();
  for (int s2 = 64; s2 > 0; s2 >>= 1) {
    if (t < s2) red[t] += red[t + s2];
    __syncthreads();
  }
  float Ssum = red[0];
  float acc = 0.f;
  for (int n = 0; n < NC; ++n) acc += expf(sA[n] - M) * hA[(size_t)n * ldh + t];
  cvec[t] = acc;
  if (t == 0) { cms[0] = M; cms[1] = Ssum; }
}

// Per-state patch over the 31-node tail; q[s] = pooled vector.
__global__ void pool_tail(const float* __restrict__ sA, const float* __restrict__ sB,
                          const float* __restrict__ hA, const float* __restrict__ hB,
                          int ldh, const float* __restrict__ cvec,
                          const float* __restrict__ cms, float* __restrict__ q) {
  int s = blockIdx.x, d = threadIdx.x;
  float CM = cms[0], CS = cms[1];
  float M = CM;
  for (int t = 0; t < 31; ++t) {
    int n = 2017 + t, shift = 30 - t;
    int bit = (s >> shift) & 1;
    float sc = bit ? sB[n] : sA[n];
    M = fmaxf(M, sc);
  }
  float scale = expf(CM - M);
  float Z = CS * scale;
  float acc = cvec[d] * scale;
  for (int t = 0; t < 31; ++t) {
    int n = 2017 + t, shift = 30 - t;
    int bit = (s >> shift) & 1;
    float sc = bit ? sB[n] : sA[n];
    float e = expf(sc - M);
    Z += e;
    const float* hr = bit ? hB : hA;
    acc += e * hr[(size_t)n * ldh + d];
  }
  q[(size_t)s * 128 + d] = acc / Z;
}

// x[rowBase+rr] = concat(q[j], ip[i]) with i = row>>8, j = row&255
__global__ void mlp_build(const float* __restrict__ q, const float* __restrict__ ip,
                          float* __restrict__ x, int rowBase, int rows) {
  int idx = blockIdx.x * 256 + threadIdx.x;
  if (idx >= rows * 256) return;
  int rr = idx >> 8, c = idx & 255;
  int grow = rowBase + rr;
  int i = grow >> 8, j = grow & 255;
  x[idx] = (c < 128) ? q[(size_t)j * 128 + c] : ip[(size_t)i * 128 + (c - 128)];
}

// LayerNorm + ReLU in place over rows of 256 (block = row)
__global__ void ln_relu(float* __restrict__ x, const float* __restrict__ g,
                        const float* __restrict__ be) {
  __shared__ float red[256];
  int row = blockIdx.x, c = threadIdx.x;
  float v = x[(size_t)row * 256 + c];
  red[c] = v;
  __syncthreads();
  for (int s2 = 128; s2 > 0; s2 >>= 1) {
    if (c < s2) red[c] += red[c + s2];
    __syncthreads();
  }
  float mean = red[0] * (1.f / 256.f);
  __syncthreads();
  float dv = v - mean;
  red[c] = dv * dv;
  __syncthreads();
  for (int s2 = 128; s2 > 0; s2 >>= 1) {
    if (c < s2) red[c] += red[c + s2];
    __syncthreads();
  }
  float var = red[0] * (1.f / 256.f);
  float o = dv * rsqrtf(var + 1e-5f) * g[c] + be[c];
  x[(size_t)row * 256 + c] = fmaxf(o, 0.f);
}

// LN + ReLU + dot(W3) + b3 -> scalar per row
__global__ void ln_relu_dot(const float* __restrict__ x, const float* __restrict__ g,
                            const float* __restrict__ be, const float* __restrict__ W3,
                            const float* __restrict__ b3, float* __restrict__ out,
                            int rowBase) {
  __shared__ float red[256];
  int row = blockIdx.x, c = threadIdx.x;
  float v = x[(size_t)row * 256 + c];
  red[c] = v;
  __syncthreads();
  for (int s2 = 128; s2 > 0; s2 >>= 1) {
    if (c < s2) red[c] += red[c + s2];
    __syncthreads();
  }
  float mean = red[0] * (1.f / 256.f);
  __syncthreads();
  float dv = v - mean;
  red[c] = dv * dv;
  __syncthreads();
  for (int s2 = 128; s2 > 0; s2 >>= 1) {
    if (c < s2) red[c] += red[c + s2];
    __syncthreads();
  }
  float var = red[0] * (1.f / 256.f);
  __syncthreads();
  float hv = fmaxf(dv * rsqrtf(var + 1e-5f) * g[c] + be[c], 0.f);
  red[c] = hv * W3[c];
  __syncthreads();
  for (int s2 = 128; s2 > 0; s2 >>= 1) {
    if (c < s2) red[c] += red[c + s2];
    __syncthreads();
  }
  if (c == 0) out[rowBase + row] = red[0] + b3[0];
}

__global__ void log_softmax_rows(const float* __restrict__ raw, float* __restrict__ out) {
  __shared__ float red[256];
  int row = blockIdx.x, c = threadIdx.x;
  float v = raw[(size_t)row * 256 + c];
  red[c] = v;
  __syncthreads();
  for (int s2 = 128; s2 > 0; s2 >>= 1) {
    if (c < s2) red[c] = fmaxf(red[c], red[c + s2]);
    __syncthreads();
  }
  float M = red[0];
  __syncthreads();
  red[c] = expf(v - M);
  __syncthreads();
  for (int s2 = 128; s2 > 0; s2 >>= 1) {
    if (c < s2) red[c] += red[c + s2];
    __syncthreads();
  }
  if (c == 0) red[0] = logf(red[0]);
  __syncthreads();
  out[(size_t)row * 256 + c] = v - M - red[0];
}

__global__ void labels_gt(const float* __restrict__ fin, const float* __restrict__ tr,
                          float* __restrict__ out) {
  int i = blockIdx.x * 256 + threadIdx.x;
  if (i < SS * SS) {
    out[SS * SS + i]     = (fin[i] > 0.f) ? 1.f : 0.f;
    out[3 * SS * SS + i] = tr[i];
  }
}

// ---------------------------------------------------------------------------
// Host orchestration
// ---------------------------------------------------------------------------
extern "C" void kernel_launch(void* const* d_in, const int* in_sizes, int n_in,
                              void* d_out, int out_size, void* d_ws, size_t ws_size,
                              hipStream_t stream) {
  (void)in_sizes; (void)n_in; (void)out_size; (void)ws_size;
  const int*   edge_index = (const int*)d_in[0];
  const float* edge_attr  = (const float*)d_in[1];
  const float* h_init0    = (const float*)d_in[2];
  const float* h_init1    = (const float*)d_in[3];
  const float* finite_mat = (const float*)d_in[4];
  const float* trans_mat  = (const float*)d_in[5];
  const float* init_W     = (const float*)d_in[6];
  const float* init_b     = (const float*)d_in[7];
  const float* aggr_W     = (const float*)d_in[8];
  const float* aggr_b     = (const float*)d_in[9];
  const float* gru_Wih    = (const float*)d_in[10];
  const float* gru_Whh    = (const float*)d_in[11];
  const float* gru_bih    = (const float*)d_in[12];
  const float* gru_bhh    = (const float*)d_in[13];
  const float* pool_W1    = (const float*)d_in[14];
  const float* pool_b1    = (const float*)d_in[15];
  const float* pool_v     = (const float*)d_in[16];
  const float* ro_W1      = (const float*)d_in[17];
  const float* ro_b1      = (const float*)d_in[18];
  const float* ro_g1      = (const float*)d_in[19];
  const float* ro_be1     = (const float*)d_in[20];
  const float* ro_W2      = (const float*)d_in[21];
  const float* ro_b2      = (const float*)d_in[22];
  const float* ro_g2      = (const float*)d_in[23];
  const float* ro_be2     = (const float*)d_in[24];
  const float* ro_W3      = (const float*)d_in[25];
  const float* ro_b3      = (const float*)d_in[26];
  float* out = (float*)d_out;

  char* wsb = (char*)d_ws;
  size_t off = 0;
  auto carve = [&](size_t bytes) -> void* {
    void* p = wsb + off;
    off = (off + bytes + 255) & ~(size_t)255;
    return p;
  };
  f16* initWt  = (f16*)carve((size_t)2 * 256 * 128 * sizeof(f16));  // nk, ldb=128
  f16* aggrWt  = (f16*)carve((size_t)4 * 256 * 288 * sizeof(f16));  // nk, ldb=288
  f16* Wih16   = (f16*)carve((size_t)4 * 768 * 256 * sizeof(f16));  // already nk
  f16* Whh16   = (f16*)carve((size_t)4 * 768 * 256 * sizeof(f16));
  f16* roW1t   = (f16*)carve((size_t)2 * 256 * 256 * sizeof(f16));  // nk, ldb=256
  f16* roW2t   = (f16*)carve((size_t)2 * 256 * 256 * sizeof(f16));
  float* h0    = (float*)carve((size_t)NN * H2 * 4);
  float* h1    = (float*)carve((size_t)NN * H2 * 4);
  float* agg   = (float*)carve((size_t)NN * H2 * 4);
  float* gi    = (float*)carve((size_t)NN * G3 * 4);
  float* gh    = (float*)carve((size_t)NN * G3 * 4);
  float* scores   = (float*)carve((size_t)8 * NN * 4);
  float* cvec     = (float*)carve((size_t)4 * 128 * 4);
  float* cms      = (float*)carve((size_t)4 * 2 * 4);
  float* qpool    = (float*)carve((size_t)4 * SS * 128 * 4);
  float* xbuf     = (float*)carve((size_t)8192 * 256 * 4);
  float* hbuf     = (float*)carve((size_t)8192 * 256 * 4);
  float* head2raw = (float*)carve((size_t)SS * SS * 4);

  dim3 blk(256);
  // weight conversion: kn f32 -> nk f16 (padded where needed)
  for (int p = 0; p < 2; ++p)
    transpose_cvt<<<(256 * 128 + 255) / 256, blk, 0, stream>>>(
        init_W + (size_t)p * 128 * 256, initWt + (size_t)p * 256 * 128, 128, 256, 128);
  for (int l = 0; l < 4; ++l)
    transpose_cvt<<<(256 * 288 + 255) / 256, blk, 0, stream>>>(
        aggr_W + (size_t)l * 272 * 256, aggrWt + (size_t)l * 256 * 288, 272, 256, 288);
  f32_to_f16_k<<<(4 * 768 * 256 + 255) / 256, blk, 0, stream>>>(gru_Wih, Wih16,
                                                                4 * 768 * 256);
  f32_to_f16_k<<<(4 * 768 * 256 + 255) / 256, blk, 0, stream>>>(gru_Whh, Whh16,
                                                                4 * 768 * 256);
  for (int p = 0; p < 2; ++p) {
    transpose_cvt<<<(256 * 256 + 255) / 256, blk, 0, stream>>>(
        ro_W1 + (size_t)p * 65536, roW1t + (size_t)p * 65536, 256, 256, 256);
    transpose_cvt<<<(256 * 256 + 255) / 256, blk, 0, stream>>>(
        ro_W2 + (size_t)p * 65536, roW2t + (size_t)p * 65536, 256, 256, 256);
  }

  // init projections (K=128, global-B path)
  gemm_wmma<128><<<dim3(H2 / 64, NN / 128), blk, 0, stream>>>(
      h_init1, DH, initWt, 128, init_b, h0, H2, NN, H2);
  gemm_wmma<128><<<dim3(H2 / 64, NN / 128), blk, 0, stream>>>(
      h_init0, DH, initWt + 256 * 128, 128, init_b + H2, h1, H2, NN, H2);

  const int* srcI = edge_index;
  const int* dstI = edge_index + EE;
  for (int r = 0; r < 4; ++r) {
    for (int l = 0; l < 4; ++l) {
      float* h = (l < 2) ? h0 : h1;
      const int* gI = (l % 2 == 0) ? srcI : dstI;
      const int* sI = (l % 2 == 0) ? dstI : srcI;
      fill_zero<<<(NN * H2 + 255) / 256, blk, 0, stream>>>(agg, NN * H2);
      aggr_wmma<<<dim3(H2 / 64, EE / 128), blk, 0, stream>>>(
          h, aggrWt + (size_t)l * 256 * 288, aggr_b + l * H2, edge_attr, gI, sI, agg);
      // GRU gate GEMMs: TDM-staged B panels (K=256, Nc=768)
      gemm_wmma_tdm<<<dim3(G3 / 64, NN / 128), blk, 0, stream>>>(
          agg, H2, Wih16 + (size_t)l * 768 * 256, gru_bih + l * G3, gi, G3, NN);
      gemm_wmma_tdm<<<dim3(G3 / 64, NN / 128), blk, 0, stream>>>(
          h, H2, Whh16 + (size_t)l * 768 * 256, gru_bhh + l * G3, gh, G3, NN);
      gru_pointwise<<<(NN * H2 + 255) / 256, blk, 0, stream>>>(h, gi, gh);
    }
  }

  // pools
  for (int p = 0; p < 4; ++p) {
    const float* hA = (p < 2) ? h0 : (h0 + DH);
    const float* hB = (p < 2) ? h1 : (h1 + DH);
    pool_score<<<NN, 128, 0, stream>>>(hA, H2, pool_W1 + p * 128 * 128,
                                       pool_b1 + p * 128, pool_v + p * 128,
                                       scores + (size_t)(2 * p) * NN);
    pool_score<<<NN, 128, 0, stream>>>(hB, H2, pool_W1 + p * 128 * 128,
                                       pool_b1 + p * 128, pool_v + p * 128,
                                       scores + (size_t)(2 * p + 1) * NN);
    pool_common<<<1, 128, 0, stream>>>(scores + (size_t)(2 * p) * NN, hA, H2,
                                       cvec + p * 128, cms + p * 2);
    pool_tail<<<SS, 128, 0, stream>>>(scores + (size_t)(2 * p) * NN,
                                      scores + (size_t)(2 * p + 1) * NN, hA, hB, H2,
                                      cvec + p * 128, cms + p * 2,
                                      qpool + (size_t)p * SS * 128);
  }

  // readout heads (MLP GEMMs use TDM-staged B panels)
  for (int hd = 0; hd < 2; ++hd) {
    const float* q  = qpool + (size_t)(hd * 2) * SS * 128;
    const float* ip = qpool + (size_t)(hd * 2 + 1) * SS * 128;
    float* target = (hd == 0) ? out : head2raw;
    for (int c = 0; c < 8; ++c) {
      int rb = c * 8192;
      mlp_build<<<(8192 * 256 + 255) / 256, blk, 0, stream>>>(q, ip, xbuf, rb, 8192);
      gemm_wmma_tdm<<<dim3(4, 64), blk, 0, stream>>>(
          xbuf, 256, roW1t + (size_t)hd * 65536, ro_b1 + hd * 256, hbuf, 256, 8192);
      ln_relu<<<8192, 256, 0, stream>>>(hbuf, ro_g1 + hd * 256, ro_be1 + hd * 256);
      gemm_wmma_tdm<<<dim3(4, 64), blk, 0, stream>>>(
          hbuf, 256, roW2t + (size_t)hd * 65536, ro_b2 + hd * 256, xbuf, 256, 8192);
      ln_relu_dot<<<8192, 256, 0, stream>>>(xbuf, ro_g2 + hd * 256, ro_be2 + hd * 256,
                                            ro_W3 + hd * 256, ro_b3 + hd, target, rb);
    }
  }

  log_softmax_rows<<<SS, 256, 0, stream>>>(head2raw, out + 2 * SS * SS);
  labels_gt<<<(SS * SS + 255) / 256, blk, 0, stream>>>(finite_mat, trans_mat, out);
}
